// RNN_88493506167243
// MI455X (gfx1250) — matrix operand order repaired and checked by reading, not compiled, not run on hardware
//
#include <hip/hip_runtime.h>
#include <stdint.h>

typedef __bf16 bf16;
typedef bf16  v16bf __attribute__((ext_vector_type(16)));
typedef float v8f   __attribute__((ext_vector_type(8)));

union Frag {
    v16bf v;
    uint4 q[2];
};

#define B_  256
#define T_  1024
#define E_  64
#define H_  256
#define V_  256
#define AK  (E_ + H_)   // 320

// workspace layout in bf16 elements
#define CW_ELEMS     (H_ * AK)          // 81920
#define HEADW_ELEMS  (V_ * H_)          // 65536
#define CW_OFF       0
#define HEADW_OFF    (CW_OFF + CW_ELEMS)
#define HIDDEN_OFF   (HEADW_OFF + HEADW_ELEMS)   // 147456 elems (16B aligned)

// ---- CDNA5 async global->LDS path (guarded so compilation can't regress) ----
#if defined(__has_builtin)
#if __has_builtin(__builtin_amdgcn_global_load_async_to_lds_b128)
#define HAVE_ASYNC_LDS 1
#endif
#if __has_builtin(__builtin_amdgcn_s_wait_asynccnt)
#define HAVE_WAIT_ASYNC_BUILTIN 1
#endif
#endif

#ifdef HAVE_ASYNC_LDS
typedef int v4i_t __attribute__((vector_size(16)));
typedef __attribute__((address_space(1))) v4i_t* as1_v4i;
typedef __attribute__((address_space(3))) v4i_t* as3_v4i;
__device__ __forceinline__ void async_copy_b128(void* lds, const void* g) {
    __builtin_amdgcn_global_load_async_to_lds_b128((as1_v4i)g, (as3_v4i)lds, 0, 0);
}
__device__ __forceinline__ void wait_async0() {
#ifdef HAVE_WAIT_ASYNC_BUILTIN
    __builtin_amdgcn_s_wait_asynccnt(0);
#else
    asm volatile("s_wait_asynccnt 0" ::: "memory");
#endif
}
#endif

// branch-free tanh: (e^{2x}-1) * rcp(e^{2x}+1), clamped so e never overflows.
__device__ __forceinline__ float fast_tanh(float x) {
    x = fminf(fmaxf(x, -10.0f), 10.0f);
    const float e = __expf(2.0f * x);                       // v_exp_f32
    return (e - 1.0f) * __builtin_amdgcn_rcpf(e + 1.0f);    // v_rcp_f32
}

// ---------------------------------------------------------------------------
// Kernel 0: convert fp32 weights -> bf16 in workspace (row-major, unchanged).
// ---------------------------------------------------------------------------
__global__ void __launch_bounds__(256) convert_weights(
    const float* __restrict__ cw, const float* __restrict__ hw,
    bf16* __restrict__ ws)
{
    int i = blockIdx.x * 256 + threadIdx.x;
    if (i < CW_ELEMS)    ws[CW_OFF + i]    = (bf16)cw[i];
    if (i < HEADW_ELEMS) ws[HEADW_OFF + i] = (bf16)hw[i];
}

// ---------------------------------------------------------------------------
// Kernel 1: RNN scan. One workgroup per 16 batch rows; 8 waves, each owning 2
// N-tiles. Weight B-fragments live in 160 VGPRs/wave; A panel [16,320] in LDS.
// Per step: commit emb_t -> barrier -> {coalesced h_{t-1} spill, emb_{t+1}
// prefetch, double-buffered WMMA} -> barrier -> tanh + h_t into LDS.
// ---------------------------------------------------------------------------
__global__ void __launch_bounds__(256) rnn_scan(
    const int*   __restrict__ x,        // [B, T]
    const float* __restrict__ start,    // [1, H]
    const float* __restrict__ wte,      // [V, E]
    const float* __restrict__ cw_b,     // [H]
    const bf16*  __restrict__ ws_cw,    // [H, AK] bf16
    bf16*        __restrict__ hidden)   // [B, T, H] bf16
{
    extern __shared__ char smem[];
    bf16* A    = (bf16*)smem;                    // 16*AK bf16 = 10240 B
    int*  toks = (int*)(smem + 16 * AK * 2);     // 16*1024 int = 64 KB

    const int tid  = threadIdx.x;
    const int lane = tid & 31;
    const int wave = tid >> 5;
    const int half = lane >> 4;
    const int nl   = lane & 15;
    const int b0   = blockIdx.x * 16;

    // ---- stage this block's token slab (rows b0..b0+15 are contiguous in x)
    {
        const uint4* src = (const uint4*)(x + (size_t)b0 * T_);
        uint4*       dst = (uint4*)toks;
#ifdef HAVE_ASYNC_LDS
        for (int i = tid; i < (16 * T_) / 4; i += 256) async_copy_b128(dst + i, src + i);
#else
        for (int i = tid; i < (16 * T_) / 4; i += 256) dst[i] = src[i];
#endif
    }

    // ---- preload weight (B-matrix) fragments into registers: 2 tiles x 10 K-blocks
    Frag bw[2][10];
#pragma unroll
    for (int tile = 0; tile < 2; ++tile) {
        const int n = (wave * 2 + tile) * 16 + nl;
        const bf16* wr = ws_cw + (size_t)n * AK + half * 8;
#pragma unroll
        for (int kb = 0; kb < 10; ++kb) {
            bw[tile][kb].q[0] = *(const uint4*)(wr + kb * 32);
            bw[tile][kb].q[1] = *(const uint4*)(wr + kb * 32 + 16);
        }
    }
    const float bias0 = cw_b[(wave * 2 + 0) * 16 + nl];
    const float bias1 = cw_b[(wave * 2 + 1) * 16 + nl];

    // ---- init hidden half of A from `start`
    for (int i = tid; i < 16 * H_; i += 256) {
        int m = i >> 8, j = i & 255;
        A[m * AK + E_ + j] = (bf16)start[j];
    }

#ifdef HAVE_ASYNC_LDS
    wait_async0();
#endif
    __syncthreads();

    // embedding pipeline registers: thread owns (row em, 4 cols at ee0)
    const int em  = tid >> 4;
    const int ee0 = (tid & 15) * 4;
    float4 f;
    {
        const int tok = toks[em * T_];
        f = *(const float4*)(wte + (size_t)tok * E_ + ee0);
    }

    // coalesced hidden-spill mapping: thread owns (row hm, 16 cols at hc0)
    const int hm  = tid >> 4;
    const int hc0 = E_ + (tid & 15) * 16;        // 16 contiguous bf16 in A's h region

    const bf16* arow = A + nl * AK + half * 8;
    const int   n0   = (wave * 2 + 0) * 16 + nl;
    const int   n1   = (wave * 2 + 1) * 16 + nl;

    for (int t = 0; t < T_; ++t) {
        // commit embedding for step t into A
        {
            union { bf16 h[4]; uint2 u; } pk;
            pk.h[0] = (bf16)f.x; pk.h[1] = (bf16)f.y;
            pk.h[2] = (bf16)f.z; pk.h[3] = (bf16)f.w;
            *(uint2*)&A[em * AK + ee0] = pk.u;
        }
        __syncthreads();   // barrier1: A = [emb_t, h_{t-1}] fully built

        // coalesced spill of h_{t-1} from LDS to global (overlaps WMMA below)
        if (t > 0) {
            const uint4* hsrc = (const uint4*)(A + hm * AK + hc0);
            uint4* gdst = (uint4*)(hidden +
                ((size_t)(b0 + hm) * T_ + (t - 1)) * H_ + (hc0 - E_));
            gdst[0] = hsrc[0];
            gdst[1] = hsrc[1];
        }

        // prefetch embedding for step t+1; overlaps the WMMA section below
        if (t + 1 < T_) {
            const int tok = toks[em * T_ + t + 1];
            f = *(const float4*)(wte + (size_t)tok * E_ + ee0);
        }

        // double-buffered A fragments: wait targets loads issued one kb earlier
        Frag a[2];
        a[0].q[0] = *(const uint4*)(arow);
        a[0].q[1] = *(const uint4*)(arow + 16);
        v8f c0 = {}, c1 = {};
#pragma unroll
        for (int kb = 0; kb < 10; ++kb) {
            if (kb + 1 < 10) {
                a[(kb + 1) & 1].q[0] = *(const uint4*)(arow + (kb + 1) * 32);
                a[(kb + 1) & 1].q[1] = *(const uint4*)(arow + (kb + 1) * 32 + 16);
            }
            c0 = __builtin_amdgcn_wmma_f32_16x16x32_bf16(false, a[kb & 1].v, false,
                     bw[0][kb].v, (short)0, c0, false, false);
            c1 = __builtin_amdgcn_wmma_f32_16x16x32_bf16(false, a[kb & 1].v, false,
                     bw[1][kb].v, (short)0, c1, false, false);
        }
        __syncthreads();   // barrier2: all waves done reading A

        // bias + tanh, write h_t into A (LDS only; global spill happens next iter)
#pragma unroll
        for (int r = 0; r < 8; ++r) {
            const int m = r + half * 8;
            A[m * AK + E_ + n0] = (bf16)fast_tanh(c0[r] + bias0);
            A[m * AK + E_ + n1] = (bf16)fast_tanh(c1[r] + bias1);
        }
    }

    __syncthreads();
    // epilogue: spill h_{T-1}
    {
        const uint4* hsrc = (const uint4*)(A + hm * AK + hc0);
        uint4* gdst = (uint4*)(hidden +
            ((size_t)(b0 + hm) * T_ + (T_ - 1)) * H_ + (hc0 - E_));
        gdst[0] = hsrc[0];
        gdst[1] = hsrc[1];
    }
}

// ---------------------------------------------------------------------------
// Kernel 2: head GEMM. out[m, v] = hidden[m, :] @ head_w[v, :] + head_b[v]
// M = 262144, N = 256, K = 256. head_w cached in 128 KB LDS per block (async
// copy when available); each of 8 waves computes a 16 x 256 strip.
// ---------------------------------------------------------------------------
__global__ void __launch_bounds__(256) head_gemm(
    const bf16*  __restrict__ hidden,   // [M, H] bf16
    const bf16*  __restrict__ headw,    // [V, H] bf16
    const float* __restrict__ head_b,   // [V]
    float*       __restrict__ out)      // [M, V]
{
    extern __shared__ bf16 W[];         // V_*H_ bf16 = 128 KB
    const int tid  = threadIdx.x;
    const int lane = tid & 31;
    const int wave = tid >> 5;
    const int half = lane >> 4;
    const int nl   = lane & 15;

    // cooperative 128-bit fill of head_w into LDS
    {
        const uint4* src = (const uint4*)headw;
        uint4*       dst = (uint4*)W;
#ifdef HAVE_ASYNC_LDS
        for (int i = tid; i < (V_ * H_) / 8; i += 256) async_copy_b128(dst + i, src + i);
        wait_async0();
#else
        for (int i = tid; i < (V_ * H_) / 8; i += 256) dst[i] = src[i];
#endif
    }
    __syncthreads();

    const size_t mBase = ((size_t)blockIdx.x * 8 + wave) * 16;

    v8f acc[16];
#pragma unroll
    for (int n = 0; n < 16; ++n) acc[n] = (v8f){};

    const bf16* arow = hidden + (mBase + nl) * H_ + half * 8;
#pragma unroll
    for (int kb = 0; kb < 8; ++kb) {
        Frag a;
        a.q[0] = *(const uint4*)(arow + kb * 32);
        a.q[1] = *(const uint4*)(arow + kb * 32 + 16);
#pragma unroll
        for (int n = 0; n < 16; ++n) {
            Frag b;
            const bf16* wr = W + (n * 16 + nl) * H_ + kb * 32 + half * 8;
            b.q[0] = *(const uint4*)(wr);
            b.q[1] = *(const uint4*)(wr + 16);
            acc[n] = __builtin_amdgcn_wmma_f32_16x16x32_bf16(false, a.v, false, b.v,
                                                             (short)0, acc[n], false, false);
        }
    }

#pragma unroll
    for (int n = 0; n < 16; ++n) {
        const int col  = n * 16 + nl;
        const float bb = head_b[col];
#pragma unroll
        for (int r = 0; r < 8; ++r) {
            const size_t m = mBase + (size_t)(r + half * 8);
            out[m * V_ + col] = acc[n][r] + bb;
        }
    }
}

// ---------------------------------------------------------------------------
extern "C" void kernel_launch(void* const* d_in, const int* in_sizes, int n_in,
                              void* d_out, int out_size, void* d_ws, size_t ws_size,
                              hipStream_t stream) {
    const int*   x     = (const int*)  d_in[0];   // [B, T]
    const float* start = (const float*)d_in[1];   // [1, H]
    const float* wte   = (const float*)d_in[2];   // [V, E]
    const float* cw_w  = (const float*)d_in[3];   // [H, E+H]
    const float* cw_b  = (const float*)d_in[4];   // [H]
    const float* hw    = (const float*)d_in[5];   // [V, H]
    const float* hb    = (const float*)d_in[6];   // [V]

    bf16* ws     = (bf16*)d_ws;
    bf16* cw_bf  = ws + CW_OFF;
    bf16* hw_bf  = ws + HEADW_OFF;
    bf16* hidden = ws + HIDDEN_OFF;               // [B*T, H] bf16, 128 MB

    // 0) weights fp32 -> bf16
    convert_weights<<<(CW_ELEMS + 255) / 256, 256, 0, stream>>>(cw_w, hw, ws);

    // 1) recurrent scan: 16 WGs x 16 batch rows, 8 waves each
    //    dynamic LDS: A panel (10240 B) + token slab (65536 B)
    rnn_scan<<<B_ / 16, 256, 16 * AK * 2 + 16 * T_ * 4, stream>>>(
        x, start, wte, cw_b, cw_bf, hidden);

    // 2) head GEMM: 2048 blocks x (8 waves * 16 rows) covers M = 262144
    head_gemm<<<(B_ * T_) / 128, 256, V_ * H_ * sizeof(bf16), stream>>>(
        hidden, hw_bf, hb, (float*)d_out);
}
